// Conv3DNorm_57440892616977
// MI455X (gfx1250) — matrix operand order repaired and checked by reading, not compiled, not run on hardware
//
#include <hip/hip_runtime.h>

typedef __attribute__((ext_vector_type(16))) __bf16 v16bf;
typedef __attribute__((ext_vector_type(4)))  __bf16 v4bf;
typedef __attribute__((ext_vector_type(8)))  float  v8f;

// ---------------------------------------------------------------------------
// Pass 1: demodulation coefficients  dcoef[cout] = rsqrt(sum(w^2) + eps)
// ---------------------------------------------------------------------------
__global__ void demod_coef_kernel(const float* __restrict__ w,
                                  float* __restrict__ dcoef) {
  __shared__ float red[256];
  const int cout = blockIdx.x;
  const float* wp = w + (size_t)cout * (128 * 27);
  float s = 0.f;
  for (int i = threadIdx.x; i < 128 * 27; i += 256) {
    float v = wp[i];
    s += v * v;
  }
  red[threadIdx.x] = s;
  __syncthreads();
  for (int off = 128; off > 0; off >>= 1) {
    if (threadIdx.x < off) red[threadIdx.x] += red[threadIdx.x + off];
    __syncthreads();
  }
  if (threadIdx.x == 0) dcoef[cout] = rsqrtf(red[0] + 1e-8f);
}

// ---------------------------------------------------------------------------
// Pass 2: bake demodulated bf16 weights into per-lane WMMA A-fragment layout.
// Fragment f = ((tap*4 + cinChunk)*8 + coutTile); 512 halves per fragment.
// A map (contiguous K, matches B): lane<16: K = j
//                                  lane>=16: K = j+16
// ---------------------------------------------------------------------------
__global__ void frag_kernel(const float* __restrict__ w,
                            const float* __restrict__ dcoef,
                            __bf16* __restrict__ fragA) {
  const int f = blockIdx.x;  // 0..863
  const int m = f & 7;
  const int c = (f >> 3) & 3;
  const int t = f >> 5;  // 0..26
  const int dw = t % 3, dy = (t / 3) % 3, dz = t / 9;
  for (int h = threadIdx.x; h < 512; h += blockDim.x) {
    const int lane = h >> 4, j = h & 15;
    int K;
    if (lane < 16) K = j;
    else           K = j + 16;
    const int cin  = c * 32 + K;
    const int cout = m * 16 + (lane & 15);
    const float v =
        w[((((size_t)cout * 128 + cin) * 3 + dz) * 3 + dy) * 3 + dw] *
        dcoef[cout];
    fragA[(size_t)f * 512 + h] = (__bf16)v;
  }
}

// ---------------------------------------------------------------------------
// Main implicit-GEMM conv kernel.
// Workgroup = (b, z, y-quad). 8 waves = 8 cout tiles of 16.
// Each wave: 8 C fragments (4 y rows x 2 half-rows of W=32).
// LDS: bf16 [6 rows][34 w][136 cin]  (w halo+pad, cin padded 128->136 for banks)
//
// Inner block per (dw, cinChunk): preload 3 dy A-fragments + 6 r B-fragments,
// then acc[yi] += A[dy] x B[yi+dy] -> 1 ds_load_b128 per WMMA, loads issued
// as a clause ahead of the MMA burst.
// ---------------------------------------------------------------------------
#define LDS_W 34
#define LDS_C 136

__global__ __launch_bounds__(256)
void conv3d_wmma_kernel(const float* __restrict__ x,
                        const __bf16* __restrict__ fragA,
                        const float* __restrict__ bias,
                        float* __restrict__ out) {
  __shared__ __align__(16) __bf16 lds[6 * LDS_W * LDS_C];

  const int tid  = threadIdx.x;
  const int lane = tid & 31;
  const int m    = tid >> 5;          // cout tile for this wave
  const int bx   = blockIdx.x;
  const int b    = bx >> 8;
  const int z    = (bx >> 3) & 31;
  const int y0   = (bx & 7) << 2;

  v8f acc[4][2] = {};

  const int n       = lane & 15;           // B/C column
  const int cinhalf = (lane >> 4) << 4;    // 0 or 16 (B K-half per lane group)

  const float* xb = x + (size_t)b * 128 * 32 * 32 * 32;

  for (int dz = 0; dz < 3; ++dz) {
    const int zz = z + dz - 1;

    // ---- cooperative LDS fill: 6 y-rows (halo) x 128 cin x 32 w, f32->bf16.
    // Quad of 4 consecutive cin per thread: 4 coalesced b32 loads, packed
    // converts, one 8-byte ds_store. Border branch is wave-uniform.
#pragma unroll 1
    for (int i = 0; i < 6144; i += 256) {
      const int q    = i + tid;
      const int wcol = q & 31;
      const int cin0 = ((q >> 5) & 31) << 2;
      const int r    = q >> 10;            // 0..5
      const int yy   = y0 + r - 1;
      v4bf pack;
      if ((unsigned)zz < 32u && (unsigned)yy < 32u) {
        const float* xp =
            xb + ((((size_t)cin0 * 32 + zz) * 32 + yy) * 32 + wcol);
        pack.x = (__bf16)xp[0];
        pack.y = (__bf16)xp[32768];
        pack.z = (__bf16)xp[65536];
        pack.w = (__bf16)xp[98304];
      } else {
        pack.x = (__bf16)0.f; pack.y = (__bf16)0.f;
        pack.z = (__bf16)0.f; pack.w = (__bf16)0.f;
      }
      *(v4bf*)&lds[(r * LDS_W + wcol + 1) * LDS_C + cin0] = pack;
    }
    // zero the w = -1 / w = 32 halo columns (vectorized by cin quads)
    {
      v4bf zq;
      zq.x = (__bf16)0.f; zq.y = (__bf16)0.f;
      zq.z = (__bf16)0.f; zq.w = (__bf16)0.f;
      for (int p = tid; p < 6 * 2 * 32; p += 256) {
        const int cin0 = (p & 31) << 2;
        const int wsel = (p >> 5) & 1;
        const int r    = p >> 6;
        *(v4bf*)&lds[(r * LDS_W + (wsel ? 33 : 0)) * LDS_C + cin0] = zq;
      }
    }
    __syncthreads();

    // ---- GEMM: for each (dw, cin chunk), 3 A-frags + 2x6 B-frags feed 24 MMAs
#pragma unroll 1
    for (int dw = 0; dw < 3; ++dw) {
#pragma unroll 1
      for (int c = 0; c < 4; ++c) {
        const int cinbase = c * 32 + cinhalf;

        v16bf a3[3];
#pragma unroll
        for (int dy = 0; dy < 3; ++dy) {
          const int f = (((dz * 3 + dy) * 3 + dw) * 4 + c) * 8 + m;
          a3[dy] = *(const v16bf*)(fragA + (size_t)f * 512 + lane * 16);
        }

#pragma unroll
        for (int wt = 0; wt < 2; ++wt) {
          const int wl = wt * 16 + n + dw;   // 0..33
          v16bf bf[6];
#pragma unroll
          for (int r = 0; r < 6; ++r)
            bf[r] = *(const v16bf*)(&lds[(r * LDS_W + wl) * LDS_C + cinbase]);
#pragma unroll
          for (int dy = 0; dy < 3; ++dy) {
#pragma unroll
            for (int yi = 0; yi < 4; ++yi) {
              acc[yi][wt] = __builtin_amdgcn_wmma_f32_16x16x32_bf16(
                  false, a3[dy], false, bf[yi + dy], (short)0, acc[yi][wt],
                  false, false);
            }
          }
        }
      }
    }
    __syncthreads();
  }

  // ---- epilogue: bias + leaky-relu + sqrt(2) gain + clamp
  const float slope = 0.2f;
  const float gain  = 1.41421356237309515f;
  const float cl    = 256.f;
  const int mrow = (lane >> 4) << 3;  // 0 or 8
#pragma unroll
  for (int yi = 0; yi < 4; ++yi) {
    const int yy = y0 + yi;
#pragma unroll
    for (int wt = 0; wt < 2; ++wt) {
      const int wcol = wt * 16 + n;
#pragma unroll
      for (int v = 0; v < 8; ++v) {
        const int cout = m * 16 + mrow + v;
        float val = acc[yi][wt][v] + bias[cout];
        val = (val > 0.f) ? val : slope * val;
        val *= gain;
        val = fminf(fmaxf(val, -cl), cl);
        out[((((size_t)b * 128 + cout) * 32 + z) * 32 + yy) * 32 + wcol] = val;
      }
    }
  }
}

// ---------------------------------------------------------------------------
extern "C" void kernel_launch(void* const* d_in, const int* in_sizes, int n_in,
                              void* d_out, int out_size, void* d_ws,
                              size_t ws_size, hipStream_t stream) {
  const float* x      = (const float*)d_in[0];
  const float* weight = (const float*)d_in[1];
  const float* bias   = (const float*)d_in[2];
  float* out          = (float*)d_out;

  float*  dcoef = (float*)d_ws;                       // 128 floats
  __bf16* fragA = (__bf16*)((char*)d_ws + 1024);      // 27*4*8*512 halves ~ 864 KB

  demod_coef_kernel<<<128, 256, 0, stream>>>(weight, dcoef);
  frag_kernel<<<27 * 4 * 8, 256, 0, stream>>>(weight, dcoef, fragA);
  conv3d_wmma_kernel<<<8 * 32 * 8, 256, 0, stream>>>(x, fragA, bias, out);
}
